// LinearAttention_55482387530451
// MI455X (gfx1250) — compile-verified, matrix operand order checked by reading, not run on hardware
//
#include <hip/hip_runtime.h>
#include <hip/hip_bf16.h>
#include <math.h>

// Problem constants (match reference setup_inputs)
#define BB 8
#define CC 128
#define NN 16384
#define SPLITS 32   // split-K blocks per batch for the KV GEMM
#define LDL 136     // padded LDS leading dim (bf16 elems), 272B rows -> 16B aligned

typedef __attribute__((ext_vector_type(16))) __bf16 v16bf;
typedef __attribute__((ext_vector_type(8)))  float  v8f;

union BfPack { uint4 u[2]; v16bf v; };

__device__ __forceinline__ unsigned short f2bf(float f) {
  union { float f; unsigned int u; } v; v.f = f;
  unsigned int r = v.u + 0x7fffu + ((v.u >> 16) & 1u);  // round-nearest-even
  return (unsigned short)(r >> 16);
}
__device__ __forceinline__ float bf2f(unsigned short u) {
  union { unsigned int u; float f; } v; v.u = ((unsigned int)u) << 16;
  return v.f;
}
__device__ __forceinline__ v8f zero8() {
  v8f z;
#pragma unroll
  for (int i = 0; i < 8; ++i) z[i] = 0.f;
  return z;
}
__device__ __forceinline__ float elu1(float z) { return z > 0.f ? z + 1.f : __expf(z); }

// Load a 16x32 bf16 fragment (16 rows of `ld`-stride memory, 32 contiguous K per row)
// into the CDNA5 WMMA 16-bit operand layout:
//   lanes 0-15  (row = lane):    VGPR0-3 = K[0..7],  VGPR4-7 = K[16..23]
//   lanes 16-31 (row = lane-16): VGPR0-3 = K[8..15], VGPR4-7 = K[24..31]
__device__ __forceinline__ v16bf load_frag(const unsigned short* tile, int ld, int lane) {
  int r = lane & 15, h = lane >> 4;
  const unsigned short* p = tile + (size_t)r * ld + h * 8;
  BfPack pk;
  pk.u[0] = *(const uint4*)(p);
  pk.u[1] = *(const uint4*)(p + 16);
  return pk.v;
}

__device__ __forceinline__ v8f wmma_bf16(v16bf a, v16bf b, v8f c) {
  return __builtin_amdgcn_wmma_f32_16x16x32_bf16(false, a, false, b, (short)0, c, false, false);
}

// ---------------- Kernel 1: BN statistics -> fold coefficients -----------------
__global__ void k_bnstats(const float* __restrict__ x, const float* __restrict__ gamma,
                          const float* __restrict__ beta, float* __restrict__ g,
                          float* __restrict__ h) {
  int c = blockIdx.x;
  float s = 0.f, s2 = 0.f;
  for (int b = 0; b < BB; ++b) {
    const float* row = x + ((size_t)b * CC + c) * NN;
    for (int n = threadIdx.x; n < NN; n += blockDim.x) {
      float v = row[n];
      s += v; s2 += v * v;
    }
  }
  __shared__ float ls[256], ls2[256];
  ls[threadIdx.x] = s; ls2[threadIdx.x] = s2;
  __syncthreads();
  for (int st = 128; st > 0; st >>= 1) {
    if (threadIdx.x < st) { ls[threadIdx.x] += ls[threadIdx.x + st]; ls2[threadIdx.x] += ls2[threadIdx.x + st]; }
    __syncthreads();
  }
  if (threadIdx.x == 0) {
    const float inv = 1.f / (float)(BB * NN);
    float mean = ls[0] * inv;
    float var  = ls2[0] * inv - mean * mean;
    float gg = gamma[c] * rsqrtf(var + 1e-5f);
    g[c] = gg;
    h[c] = beta[c] - mean * gg;
  }
}

// ------------- Kernel 2: fold BN into weights: W' bf16, effective bias ----------
__global__ void k_fold(const float* __restrict__ Wq, const float* __restrict__ bq,
                       const float* __restrict__ Wk, const float* __restrict__ bk,
                       const float* __restrict__ Wv, const float* __restrict__ bv,
                       const float* __restrict__ g, const float* __restrict__ h,
                       unsigned short* __restrict__ Wqkv, float* __restrict__ bqkv) {
  int o = blockIdx.x;           // 0..383
  int c = threadIdx.x;          // 0..127
  const float* W; const float* bi; int orow = o;
  if (o < 128)      { W = Wq; bi = bq; }
  else if (o < 256) { W = Wk; bi = bk; orow = o - 128; }
  else              { W = Wv; bi = bv; orow = o - 256; }
  float w = W[orow * CC + c];
  Wqkv[o * CC + c] = f2bf(w * g[c]);
  __shared__ float red[128];
  red[c] = w * h[c];
  __syncthreads();
  for (int st = 64; st > 0; st >>= 1) {
    if (c < st) red[c] += red[c + st];
    __syncthreads();
  }
  if (c == 0) bqkv[o] = bi[orow] + red[0];
}

// One 16-row output strip of the fused QKV GEMM. `act` is a compile-time flag
// after inlining -> completely branch-free epilogue, single-base b16 stores.
__device__ __forceinline__ void qkv_strip(const unsigned short* __restrict__ xl,
                                          const unsigned short* __restrict__ Wqkv,
                                          const float* __restrict__ bqkv,
                                          unsigned short* __restrict__ dst,
                                          int b, int n0, int seg0,
                                          int wave, int lane, int h, int cl, bool act) {
  const float scale = 0.08838834764831845f;  // C^-0.5
  int m0g = seg0 + wave * 16;                 // global row base in [0,384)
  v16bf a[4];
#pragma unroll
  for (int k = 0; k < 4; ++k)
    a[k] = load_frag(Wqkv + (size_t)m0g * CC + k * 32, CC, lane);
  float bias[8];
#pragma unroll
  for (int r = 0; r < 8; ++r) bias[r] = bqkv[m0g + r + h * 8];

#pragma unroll
  for (int nt = 0; nt < 8; ++nt) {
    v8f acc = zero8();
#pragma unroll
    for (int k = 0; k < 4; ++k)
      acc = wmma_bf16(a[k], load_frag(xl + nt * 16 * LDL + k * 32, LDL, lane), acc);
    int col = n0 + nt * 16 + cl;
    int row_l = wave * 16 + h * 8;            // local row in destination buffer
    unsigned short* o = dst + ((size_t)b * CC + row_l) * NN + col;
#pragma unroll
    for (int r = 0; r < 8; ++r) {
      float v = acc[r] + bias[r];
      if (act) v = elu1(v * scale);
      o[(size_t)r * NN] = f2bf(v);
    }
  }
}

// ---- Kernel 3: fused QKV projection GEMM [384x128]x[128x128-tile] + activation ----
__global__ void k_gemm_qkv(const float* __restrict__ x, const unsigned short* __restrict__ Wqkv,
                           const float* __restrict__ bqkv,
                           unsigned short* __restrict__ Qb, unsigned short* __restrict__ Kb,
                           unsigned short* __restrict__ Vb) {
  __shared__ unsigned short xl[128 * LDL];   // x tile, stored [n][c] (N-major) bf16
  const int tiles_n = NN / 128;
  int b  = blockIdx.x / tiles_n;
  int n0 = (blockIdx.x % tiles_n) * 128;
  int tid = threadIdx.x, lane = tid & 31, wave = tid >> 5;

  // stage + transpose + f32->bf16, channel rows paired -> packed b32 DS writes
  for (int idx = tid; idx < 64 * 128; idx += 256) {
    int c2 = idx >> 7, n = idx & 127;
    float v0 = x[((size_t)b * CC + 2 * c2 + 0) * NN + n0 + n];
    float v1 = x[((size_t)b * CC + 2 * c2 + 1) * NN + n0 + n];
    unsigned int pk = (unsigned int)f2bf(v0) | ((unsigned int)f2bf(v1) << 16);
    *(unsigned int*)&xl[n * LDL + 2 * c2] = pk;
  }
  __syncthreads();

  int h = lane >> 4, cl = lane & 15;
  qkv_strip(xl, Wqkv, bqkv, Qb, b, n0, 0,   wave, lane, h, cl, true);
  qkv_strip(xl, Wqkv, bqkv, Kb, b, n0, 128, wave, lane, h, cl, true);
  qkv_strip(xl, Wqkv, bqkv, Vb, b, n0, 256, wave, lane, h, cl, false);
}

// ---------------- Kernel 4: zero f32 accumulators (KV + Ksum) ------------------
__global__ void k_zero(float* __restrict__ p, int count) {
  int i = blockIdx.x * blockDim.x + threadIdx.x;
  if (i < count) p[i] = 0.f;
}

// ---------------- Kernel 5: K row sums over N ----------------------------------
__global__ void k_ksum(const unsigned short* __restrict__ Kb, float* __restrict__ Ksum) {
  int b = blockIdx.x / CC, c = blockIdx.x % CC;
  const unsigned short* row = Kb + ((size_t)b * CC + c) * NN;
  float s = 0.f;
  for (int n = threadIdx.x; n < NN; n += blockDim.x) s += bf2f(row[n]);
  __shared__ float red[256];
  red[threadIdx.x] = s;
  __syncthreads();
  for (int st = 128; st > 0; st >>= 1) {
    if (threadIdx.x < st) red[threadIdx.x] += red[threadIdx.x + st];
    __syncthreads();
  }
  if (threadIdx.x == 0) Ksum[b * CC + c] = red[0];
}

// -------- Kernel 6: KV[b,v,k] = sum_n V[b,v,n] K[b,k,n], split-K + atomics ------
__global__ void k_gemm_kv(const unsigned short* __restrict__ Vb,
                          const unsigned short* __restrict__ Kb,
                          float* __restrict__ KV) {
  int b = blockIdx.x / SPLITS;
  int split = blockIdx.x % SPLITS;
  int tid = threadIdx.x, lane = tid & 31, wave = tid >> 5;
  int m0 = wave * 16;                       // c_v strip
  const int span = NN / SPLITS;             // 512 n per split
  int kbeg = split * span;

  v8f acc[8];
#pragma unroll
  for (int nt = 0; nt < 8; ++nt) acc[nt] = zero8();

  for (int k0 = kbeg; k0 < kbeg + span; k0 += 32) {
    v16bf a = load_frag(Vb + ((size_t)b * CC + m0) * NN + k0, NN, lane);
#pragma unroll
    for (int nt = 0; nt < 8; ++nt) {
      v16bf bf = load_frag(Kb + ((size_t)b * CC + nt * 16) * NN + k0, NN, lane);
      acc[nt] = wmma_bf16(a, bf, acc[nt]);
    }
  }
  int h = lane >> 4, cl = lane & 15;
#pragma unroll
  for (int nt = 0; nt < 8; ++nt) {
    float* base = &KV[((size_t)b * CC + m0 + h * 8) * CC + nt * 16 + cl];
#pragma unroll
    for (int r = 0; r < 8; ++r)
      atomicAdd(base + (size_t)r * CC, acc[nt][r]);
  }
}

// ---------------- Kernel 7: KV f32 -> bf16 -------------------------------------
__global__ void k_kvcast(const float* __restrict__ KV, unsigned short* __restrict__ KVb, int count) {
  int i = blockIdx.x * blockDim.x + threadIdx.x;
  if (i < count) KVb[i] = f2bf(KV[i]);
}

// ---- Kernel 8: out = (KV . Q) / max(Q.Ksum, eps) + x --------------------------
__global__ void k_gemm_out(const unsigned short* __restrict__ Qb,
                           const unsigned short* __restrict__ KVb,
                           const float* __restrict__ Ksum,
                           const float* __restrict__ x, float* __restrict__ out) {
  __shared__ unsigned short ql[128 * LDL];  // Q tile as [n][k] bf16
  __shared__ float ksum_l[128];
  __shared__ float denom_l[128];
  const int tiles_n = NN / 128;
  int b  = blockIdx.x / tiles_n;
  int n0 = (blockIdx.x % tiles_n) * 128;
  int tid = threadIdx.x, lane = tid & 31, wave = tid >> 5;

  if (tid < 128) ksum_l[tid] = Ksum[b * CC + tid];
  // stage Q tile transposed; k rows paired -> packed b32 DS writes
  for (int idx = tid; idx < 64 * 128; idx += 256) {
    int k2 = idx >> 7, n = idx & 127;
    unsigned int lo = Qb[((size_t)b * CC + 2 * k2 + 0) * NN + n0 + n];
    unsigned int hi = Qb[((size_t)b * CC + 2 * k2 + 1) * NN + n0 + n];
    *(unsigned int*)&ql[n * LDL + 2 * k2] = lo | (hi << 16);
  }
  __syncthreads();
  if (tid < 128) {
    float d = 0.f;
#pragma unroll 8
    for (int k = 0; k < 128; ++k) d += bf2f(ql[tid * LDL + k]) * ksum_l[k];
    denom_l[tid] = fmaxf(d, 1e-4f);
  }
  __syncthreads();

  int m0 = wave * 16;                       // c_v strip (8 waves cover 128 rows)
  v16bf a[4];
#pragma unroll
  for (int k = 0; k < 4; ++k)
    a[k] = load_frag(KVb + (size_t)b * CC * CC + m0 * CC + k * 32, CC, lane);

  int h = lane >> 4, cl = lane & 15;
#pragma unroll
  for (int nt = 0; nt < 8; ++nt) {
    v8f acc = zero8();
#pragma unroll
    for (int k = 0; k < 4; ++k)
      acc = wmma_bf16(a[k], load_frag(ql + nt * 16 * LDL + k * 32, LDL, lane), acc);
    int col_l = nt * 16 + cl;
    int col = n0 + col_l;
    float d = denom_l[col_l];
    size_t base = ((size_t)b * CC + m0 + h * 8) * NN + col;
#pragma unroll
    for (int r = 0; r < 8; ++r) {
      size_t idx = base + (size_t)r * NN;
      out[idx] = acc[r] / d + x[idx];
    }
  }
}

// ------------------------------- launcher --------------------------------------
extern "C" void kernel_launch(void* const* d_in, const int* in_sizes, int n_in,
                              void* d_out, int out_size, void* d_ws, size_t ws_size,
                              hipStream_t stream) {
  const float* x     = (const float*)d_in[0];
  const float* Wq    = (const float*)d_in[1];
  const float* bq    = (const float*)d_in[2];
  const float* Wk    = (const float*)d_in[3];
  const float* bk    = (const float*)d_in[4];
  const float* Wv    = (const float*)d_in[5];
  const float* bv    = (const float*)d_in[6];
  const float* gamma = (const float*)d_in[7];
  const float* beta  = (const float*)d_in[8];
  float* out = (float*)d_out;

  unsigned char* ws = (unsigned char*)d_ws;
  const size_t QKV_BYTES = (size_t)BB * CC * NN * 2;       // 32 MiB each
  size_t off = 0;
  float*          g_co  = (float*)(ws + off);        off += 512;
  float*          h_co  = (float*)(ws + off);        off += 512;
  unsigned short* Wqkv  = (unsigned short*)(ws + off); off += 384 * CC * 2;   // 98304
  float*          bqkv  = (float*)(ws + off);        off += 384 * 4;
  off = (off + 255) & ~(size_t)255;
  unsigned short* Qb    = (unsigned short*)(ws + off); off += QKV_BYTES;
  unsigned short* Kb    = (unsigned short*)(ws + off); off += QKV_BYTES;
  unsigned short* Vb    = (unsigned short*)(ws + off); off += QKV_BYTES;
  float*          KV    = (float*)(ws + off);        off += (size_t)BB * CC * CC * 4;
  float*          Ksum  = (float*)(ws + off);        off += BB * CC * 4;   // contiguous after KV
  unsigned short* KVb   = (unsigned short*)(ws + off); off += (size_t)BB * CC * CC * 2;
  (void)ws_size; (void)in_sizes; (void)n_in; (void)out_size;

  k_bnstats<<<CC, 256, 0, stream>>>(x, gamma, beta, g_co, h_co);
  k_fold<<<384, 128, 0, stream>>>(Wq, bq, Wk, bk, Wv, bv, g_co, h_co, Wqkv, bqkv);
  k_gemm_qkv<<<BB * (NN / 128), 256, 0, stream>>>(x, Wqkv, bqkv, Qb, Kb, Vb);
  int zcount = BB * CC * CC + BB * CC;                     // KV + Ksum (contiguous)
  k_zero<<<(zcount + 255) / 256, 256, 0, stream>>>(KV, zcount);
  k_ksum<<<BB * CC, 256, 0, stream>>>(Kb, Ksum);
  k_gemm_kv<<<BB * SPLITS, 256, 0, stream>>>(Vb, Kb, KV);
  k_kvcast<<<(BB * CC * CC + 255) / 256, 256, 0, stream>>>(KV, KVb, BB * CC * CC);
  k_gemm_out<<<BB * (NN / 128), 256, 0, stream>>>(Qb, KVb, Ksum, x, out);
}
